// TransformerEncoderLayer_61984968016132
// MI455X (gfx1250) — matrix operand order repaired and checked by reading, not compiled, not run on hardware
//
#include <hip/hip_runtime.h>
#include <cstdint>

// Problem constants (from reference)
#define BB 8
#define LL 2048
#define EE 1024
#define NH 16
#define DD 64
#define FF 4096   // 4*E

typedef __attribute__((ext_vector_type(16))) __bf16 v16bf;
typedef __attribute__((ext_vector_type(8)))  float  v8f;

union Frag   { v16bf v; uint4 u[2]; };
union Pack16 { uint4 u[2]; __bf16 h[16]; };

__device__ __forceinline__ __bf16 f2bf(float x) { return (__bf16)x; }
__device__ __forceinline__ float  bf2f(__bf16 x) { return (float)x; }

// ---------------------------------------------------------------------------
// CDNA5 async global->LDS staging (ASYNCcnt-tracked, ISA 15.18.3 op 98)
// dst operand = VGPR holding LDS byte address; src = VGPR pair global address.
// ---------------------------------------------------------------------------
__device__ __forceinline__ unsigned lds_off(const void* p) {
    // LDS aperture: flat addr [31:0] == LDS byte offset (ISA 10.2)
    return (unsigned)(uintptr_t)p;
}
__device__ __forceinline__ void async_load_b128(unsigned lds, const void* gaddr) {
    asm volatile("global_load_async_to_lds_b128 %0, %1, off"
                 :: "v"(lds), "v"(gaddr) : "memory");
}
__device__ __forceinline__ void wait_async_le4() {
    asm volatile("s_wait_asynccnt 0x4" ::: "memory");
}
__device__ __forceinline__ void wait_async_0() {
    asm volatile("s_wait_asynccnt 0x0" ::: "memory");
}

// ---------------------------------------------------------------------------
// f32 -> bf16 conversion with transpose: in = W[N][K] row-major (torch layout),
// out = Wt[K][N] bf16. Consecutive threads -> consecutive n (coalesced writes).
// ---------------------------------------------------------------------------
__global__ __launch_bounds__(256)
void cvt_transpose_kernel(const float* __restrict__ in, __bf16* __restrict__ out,
                          int Nn, int Kk) {
    int i = blockIdx.x * 256 + threadIdx.x;
    if (i >= Nn * Kk) return;
    int k = i / Nn, n = i - k * Nn;
    out[i] = f2bf(in[(size_t)n * Kk + k]);
}

// ---------------------------------------------------------------------------
// LayerNorm (1024 cols) -> bf16
// ---------------------------------------------------------------------------
__global__ __launch_bounds__(256)
void ln_bf16_kernel(const float* __restrict__ x, const float* __restrict__ g,
                    const float* __restrict__ bta, __bf16* __restrict__ out) {
    __shared__ float red[256];
    const int row = blockIdx.x;
    const int tid = threadIdx.x;
    const float* xr = x + (size_t)row * EE;
    float4 v = ((const float4*)xr)[tid];

    red[tid] = v.x + v.y + v.z + v.w;
    __syncthreads();
    for (int off = 128; off > 0; off >>= 1) {
        if (tid < off) red[tid] += red[tid + off];
        __syncthreads();
    }
    const float mean = red[0] * (1.0f / EE);
    __syncthreads();

    float dx = v.x - mean, dy = v.y - mean, dz = v.z - mean, dw = v.w - mean;
    red[tid] = dx * dx + dy * dy + dz * dz + dw * dw;
    __syncthreads();
    for (int off = 128; off > 0; off >>= 1) {
        if (tid < off) red[tid] += red[tid + off];
        __syncthreads();
    }
    const float inv = rsqrtf(red[0] * (1.0f / EE) + 1e-5f);

    float4 gg = ((const float4*)g)[tid];
    float4 bb = ((const float4*)bta)[tid];
    __bf16* o = out + (size_t)row * EE + tid * 4;
    o[0] = f2bf(dx * inv * gg.x + bb.x);
    o[1] = f2bf(dy * inv * gg.y + bb.y);
    o[2] = f2bf(dz * inv * gg.z + bb.z);
    o[3] = f2bf(dw * inv * gg.w + bb.w);
}

// ---------------------------------------------------------------------------
// WMMA GEMM: C[M,N] = act(A[M,K] @ Wt[K,N] + bias) (+ resid)
// 256 threads = 8 waves; block tile 128(M) x 128(N); K-step 32.
// Wave grid 4(M) x 2(N): each wave 32x64 = 8 WMMA tiles, 64 accum VGPRs.
// Double-buffered LDS staged with async global->LDS b128 (ASYNCcnt).
// ---------------------------------------------------------------------------
enum { ACT_NONE = 0, ACT_ELU1 = 1, ACT_RELU = 2 };

template <int ACT, int RESID, int OUTBF>
__global__ __launch_bounds__(256)
void gemm_wmma_kernel(const __bf16* __restrict__ A, int lda,
                      const __bf16* __restrict__ Wt,   // K x N bf16, k-major
                      const float* __restrict__ bias,
                      const float* __restrict__ resid,
                      void* __restrict__ out, int N, int K) {
    __shared__ __bf16 sA[2][128][40];   // m-major (contiguous k per row)
    __shared__ __bf16 sB[2][32][136];   // k-major (contiguous n per row)

    const int tid  = threadIdx.x;
    const int wid  = tid >> 5;
    const int lane = tid & 31;
    const int gm0  = blockIdx.y * 128;
    const int gn0  = blockIdx.x * 128;
    const int wm0  = (wid >> 1) * 32;   // wave M offset in tile
    const int wn0  = (wid & 1) * 64;    // wave N offset in tile

    // per-thread staging coordinates (4 x b128 per thread per K-step)
    const int arow  = tid >> 1;          // 0..127
    const int ahalf = (tid & 1) * 16;    // 0/16
    const int brow  = tid >> 3;          // 0..31  (k)
    const int bcol  = (tid & 7) * 16;    // 0..112 (n)

    auto stage = [&](int buf, int k0) {
        const __bf16* asrc = A + (size_t)(gm0 + arow) * lda + k0 + ahalf;
        async_load_b128(lds_off(&sA[buf][arow][ahalf]),     asrc);
        async_load_b128(lds_off(&sA[buf][arow][ahalf + 8]), asrc + 8);
        const __bf16* bsrc = Wt + (size_t)(k0 + brow) * N + gn0 + bcol;
        async_load_b128(lds_off(&sB[buf][brow][bcol]),     bsrc);
        async_load_b128(lds_off(&sB[buf][brow][bcol + 8]), bsrc + 8);
    };

    v8f acc[2][4] = {};

    stage(0, 0);
    for (int k0 = 0; k0 < K; k0 += 32) {
        const int cur = (k0 >> 5) & 1;
        const bool more = (k0 + 32) < K;
        if (more) stage(cur ^ 1, k0 + 32);       // prefetch next tile
        if (more) wait_async_le4(); else wait_async_0();  // current tile done
        __syncthreads();

        // A fragment (16x32 bf16): lanes 0-15 hold K{0..7,16..23} of row
        // M=lane&15; lanes 16-31 hold K{8..15,24..31}.
        const int ksel = (lane >> 4) * 8;
        Frag fa[2];
        #pragma unroll
        for (int i = 0; i < 2; ++i) {
            const int mrow = wm0 + i * 16 + (lane & 15);
            fa[i].u[0] = *(const uint4*)&sA[cur][mrow][ksel];
            fa[i].u[1] = *(const uint4*)&sA[cur][mrow][16 + ksel];
        }
        #pragma unroll
        for (int j = 0; j < 4; ++j) {
            // B fragment (32x16): lane l holds column n=l&15, K{0..7,16..23} (lanes 0-15) or K{8..15,24..31} (lanes 16-31).
            Frag fb;
            #pragma unroll
            for (int s = 0; s < 16; ++s)
                fb.v[s] = sB[cur][(s & 7) + ksel + (s >> 3) * 16][wn0 + j * 16 + (lane & 15)];
            #pragma unroll
            for (int i = 0; i < 2; ++i)
                acc[i][j] = __builtin_amdgcn_wmma_f32_16x16x32_bf16(
                    false, fa[i].v, false, fb.v, (short)0, acc[i][j], false, false);
        }
        __syncthreads();   // done reading 'cur' before it is overwritten
    }

    // Epilogue. C/D: VGPR r, lanes0-15 -> M=r, lanes16-31 -> M=8+r; N=lane&15.
    const int nl   = lane & 15;
    const int msel = (lane >> 4) * 8;
    #pragma unroll
    for (int j = 0; j < 4; ++j) {
        const int n = gn0 + wn0 + j * 16 + nl;
        const float bn = bias[n];
        #pragma unroll
        for (int i = 0; i < 2; ++i) {
            #pragma unroll
            for (int r = 0; r < 8; ++r) {
                const int m = gm0 + wm0 + i * 16 + msel + r;
                float c = acc[i][j][r] + bn;
                if (ACT == ACT_ELU1) c = (c > 0.0f) ? (c + 1.0f) : expf(c);
                if (ACT == ACT_RELU) c = fmaxf(c, 0.0f);
                if (RESID) c += resid[(size_t)m * N + n];
                if (OUTBF) ((__bf16*)out)[(size_t)m * N + n] = f2bf(c);
                else       ((float*)out)[(size_t)m * N + n] = c;
            }
        }
    }
}

// ---------------------------------------------------------------------------
// KV[bh][d][m] = sum_l Kf[b,l,h,d] * V[b,l,h,m]   (M=64 d, N=64 m, K=2048 l)
// ---------------------------------------------------------------------------
__global__ __launch_bounds__(128)
void kv_wmma_kernel(const __bf16* __restrict__ Kf, const __bf16* __restrict__ V,
                    __bf16* __restrict__ KV) {
    __shared__ __bf16 lK[64][40];  // [d][l_local]  (transposed K tile)
    __shared__ __bf16 lV[32][72];  // [l_local][m]

    const int bh = blockIdx.x;
    const int b = bh >> 4, h = bh & 15;
    const int tid = threadIdx.x;
    const int wid = tid >> 5, lane = tid & 31;
    const int li = tid >> 2;             // 0..31 : l row within tile
    const int part = (tid & 3) * 16;     // 16-wide d/m chunk

    v8f acc[4] = {};
    for (int l0 = 0; l0 < LL; l0 += 32) {
        __syncthreads();
        const size_t rowbase = ((size_t)(b * LL + l0 + li)) * EE + h * 64 + part;
        {   // K tile: transpose-scatter to d-major
            Pack16 p;
            p.u[0] = ((const uint4*)(Kf + rowbase))[0];
            p.u[1] = ((const uint4*)(Kf + rowbase))[1];
            #pragma unroll
            for (int i = 0; i < 16; ++i) lK[part + i][li] = p.h[i];
        }
        {   // V tile: direct row-major
            *(uint4*)&lV[li][part]     = ((const uint4*)(V + rowbase))[0];
            *(uint4*)&lV[li][part + 8] = ((const uint4*)(V + rowbase))[1];
        }
        __syncthreads();

        const int drow = wid * 16 + (lane & 15);
        const int ksel = (lane >> 4) * 8;
        Frag fa;
        fa.u[0] = *(const uint4*)&lK[drow][ksel];
        fa.u[1] = *(const uint4*)&lK[drow][16 + ksel];
        #pragma unroll
        for (int j = 0; j < 4; ++j) {
            Frag fb;
            #pragma unroll
            for (int s = 0; s < 16; ++s)
                fb.v[s] = lV[(s & 7) + ksel + (s >> 3) * 16][j * 16 + (lane & 15)];
            acc[j] = __builtin_amdgcn_wmma_f32_16x16x32_bf16(
                false, fa.v, false, fb.v, (short)0, acc[j], false, false);
        }
    }

    const int nl = lane & 15, msel = (lane >> 4) * 8;
    #pragma unroll
    for (int j = 0; j < 4; ++j)
        #pragma unroll
        for (int r = 0; r < 8; ++r) {
            const int d = wid * 16 + msel + r;
            KV[(size_t)bh * 4096 + d * 64 + j * 16 + nl] = f2bf(acc[j][r]);
        }
}

// ---------------------------------------------------------------------------
// Ksum[bh][d] = sum_l Kf[b,l,h,d]
// ---------------------------------------------------------------------------
__global__ __launch_bounds__(64)
void ksum_kernel(const __bf16* __restrict__ Kf, float* __restrict__ Ksum) {
    const int bh = blockIdx.x;
    const int b = bh >> 4, h = bh & 15;
    const int d = threadIdx.x;
    float s = 0.0f;
    for (int l = 0; l < LL; ++l)
        s += bf2f(Kf[((size_t)(b * LL + l)) * EE + h * 64 + d]);
    Ksum[bh * 64 + d] = s;
}

// ---------------------------------------------------------------------------
// Z[b,l,h] = 1 / (Qf[b,l,h,:] . Ksum[b,h,:] + eps)
// ---------------------------------------------------------------------------
__global__ __launch_bounds__(256)
void z_kernel(const __bf16* __restrict__ Qf, const float* __restrict__ Ksum,
              float* __restrict__ Zv, int total) {
    int idx = blockIdx.x * 256 + threadIdx.x;
    if (idx >= total) return;
    const int h = idx & (NH - 1);
    const int bl = idx / NH;
    const int b = bl / LL;
    const __bf16* q = Qf + (size_t)bl * EE + h * 64;
    const float* ks = Ksum + (b * NH + h) * 64;
    float s = 0.0f;
    #pragma unroll 8
    for (int d = 0; d < 64; ++d) s += bf2f(q[d]) * ks[d];
    Zv[idx] = 1.0f / (s + 1e-6f);
}

// ---------------------------------------------------------------------------
// attn[b,l,h,m] = (Qf[b,l,h,:] @ KV[bh][:,m]) * Z[b,l,h]
// grid: (L/64, B*H); M=64 l, N=64 m, K=64 d
// ---------------------------------------------------------------------------
__global__ __launch_bounds__(128)
void attn_wmma_kernel(const __bf16* __restrict__ Qf, const __bf16* __restrict__ KV,
                      const float* __restrict__ Zv, __bf16* __restrict__ Attn) {
    __shared__ __bf16 lQ[64][72];   // [l][d]
    __shared__ __bf16 lKV[64][72];  // [d][m]

    const int bh = blockIdx.y;
    const int b = bh >> 4, h = bh & 15;
    const int l0 = blockIdx.x * 64;
    const int tid = threadIdx.x, wid = tid >> 5, lane = tid & 31;

    for (int u = tid; u < 512; u += 128) {
        const int row = u >> 3, c = (u & 7) * 8;
        *(uint4*)&lQ[row][c] =
            *(const uint4*)(Qf + ((size_t)(b * LL + l0 + row)) * EE + h * 64 + c);
        *(uint4*)&lKV[row][c] = *(const uint4*)(KV + (size_t)bh * 4096 + row * 64 + c);
    }
    __syncthreads();

    v8f acc[4] = {};
    #pragma unroll
    for (int d0 = 0; d0 < 64; d0 += 32) {
        const int mrow = wid * 16 + (lane & 15);
        const int ksel = (lane >> 4) * 8;
        Frag fa;
        fa.u[0] = *(const uint4*)&lQ[mrow][d0 + ksel];
        fa.u[1] = *(const uint4*)&lQ[mrow][d0 + 16 + ksel];
        #pragma unroll
        for (int j = 0; j < 4; ++j) {
            Frag fb;
            #pragma unroll
            for (int s = 0; s < 16; ++s)
                fb.v[s] = lKV[d0 + (s & 7) + ksel + (s >> 3) * 16][j * 16 + (lane & 15)];
            acc[j] = __builtin_amdgcn_wmma_f32_16x16x32_bf16(
                false, fa.v, false, fb.v, (short)0, acc[j], false, false);
        }
    }

    const int nl = lane & 15, msel = (lane >> 4) * 8;
    float zr[8];
    #pragma unroll
    for (int r = 0; r < 8; ++r) {
        const int l = l0 + wid * 16 + msel + r;
        zr[r] = Zv[((size_t)(b * LL + l)) * NH + h];
    }
    #pragma unroll
    for (int j = 0; j < 4; ++j)
        #pragma unroll
        for (int r = 0; r < 8; ++r) {
            const int l = l0 + wid * 16 + msel + r;
            Attn[((size_t)(b * LL + l)) * EE + h * 64 + j * 16 + nl] =
                f2bf(acc[j][r] * zr[r]);
        }
}

// ---------------------------------------------------------------------------
// Host-side orchestration
// ---------------------------------------------------------------------------
extern "C" void kernel_launch(void* const* d_in, const int* in_sizes, int n_in,
                              void* d_out, int out_size, void* d_ws, size_t ws_size,
                              hipStream_t stream) {
    (void)in_sizes; (void)n_in; (void)out_size;

    const float* x    = (const float*)d_in[0];
    const float* ln1g = (const float*)d_in[1];
    const float* ln1b = (const float*)d_in[2];
    const float* ln2g = (const float*)d_in[3];
    const float* ln2b = (const float*)d_in[4];
    const float* Wq = (const float*)d_in[5];  const float* bq = (const float*)d_in[6];
    const float* Wk = (const float*)d_in[7];  const float* bk = (const float*)d_in[8];
    const float* Wv = (const float*)d_in[9];  const float* bv = (const float*)d_in[10];
    const float* Wo = (const float*)d_in[11]; const float* bo = (const float*)d_in[12];
    const float* W1 = (const float*)d_in[13]; const float* b1 = (const float*)d_in[14];
    const float* W2 = (const float*)d_in[15]; const float* b2 = (const float*)d_in[16];

    const size_t TOK = (size_t)BB * LL;  // 16384 tokens

    char* p = (char*)d_ws;
    auto alloc = [&](size_t bytes) {
        void* r = (void*)p;
        p += (bytes + 255) & ~(size_t)255;
        return r;
    };
    __bf16* hbf  = (__bf16*)alloc(TOK * EE * 2);
    __bf16* WqT  = (__bf16*)alloc((size_t)EE * EE * 2);   // [K][N] transposed bf16
    __bf16* WkT  = (__bf16*)alloc((size_t)EE * EE * 2);
    __bf16* WvT  = (__bf16*)alloc((size_t)EE * EE * 2);
    __bf16* WoT  = (__bf16*)alloc((size_t)EE * EE * 2);
    __bf16* W1T  = (__bf16*)alloc((size_t)FF * EE * 2);   // [1024][4096]
    __bf16* W2T  = (__bf16*)alloc((size_t)EE * FF * 2);   // [4096][1024]
    __bf16* Qf   = (__bf16*)alloc(TOK * EE * 2);
    __bf16* Kf   = (__bf16*)alloc(TOK * EE * 2);
    __bf16* Vb   = (__bf16*)alloc(TOK * EE * 2);
    __bf16* KVb  = (__bf16*)alloc((size_t)BB * NH * DD * DD * 2);
    float*  Ksum = (float*) alloc((size_t)BB * NH * DD * 4);
    float*  Zv   = (float*) alloc(TOK * NH * 4);
    __bf16* Attn = (__bf16*)alloc(TOK * EE * 2);
    float*  x2   = (float*) alloc(TOK * EE * 4);
    __bf16* h2bf = (__bf16*)alloc(TOK * EE * 2);
    __bf16* h1bf = (__bf16*)alloc(TOK * FF * 2);
    if ((size_t)(p - (char*)d_ws) > ws_size) return;  // workspace too small

    // 1. weights -> transposed bf16 (k-major, async/WMMA-friendly)
    const int nE2 = EE * EE, nF = FF * EE;
    cvt_transpose_kernel<<<(nE2 + 255) / 256, 256, 0, stream>>>(Wq, WqT, EE, EE);
    cvt_transpose_kernel<<<(nE2 + 255) / 256, 256, 0, stream>>>(Wk, WkT, EE, EE);
    cvt_transpose_kernel<<<(nE2 + 255) / 256, 256, 0, stream>>>(Wv, WvT, EE, EE);
    cvt_transpose_kernel<<<(nE2 + 255) / 256, 256, 0, stream>>>(Wo, WoT, EE, EE);
    cvt_transpose_kernel<<<(nF + 255) / 256, 256, 0, stream>>>(W1, W1T, FF, EE);
    cvt_transpose_kernel<<<(nF + 255) / 256, 256, 0, stream>>>(W2, W2T, EE, FF);

    // 2. LN1
    ln_bf16_kernel<<<(int)TOK, 256, 0, stream>>>(x, ln1g, ln1b, hbf);

    // 3. Q/K/V projections (Q,K with fused elu+1)
    dim3 gE(EE / 128, (int)(TOK / 128));
    gemm_wmma_kernel<ACT_ELU1, 0, 1><<<gE, 256, 0, stream>>>(hbf, EE, WqT, bq, nullptr, Qf, EE, EE);
    gemm_wmma_kernel<ACT_ELU1, 0, 1><<<gE, 256, 0, stream>>>(hbf, EE, WkT, bk, nullptr, Kf, EE, EE);
    gemm_wmma_kernel<ACT_NONE, 0, 1><<<gE, 256, 0, stream>>>(hbf, EE, WvT, bv, nullptr, Vb, EE, EE);

    // 4. linear attention state
    ksum_kernel<<<BB * NH, 64, 0, stream>>>(Kf, Ksum);
    kv_wmma_kernel<<<BB * NH, 128, 0, stream>>>(Kf, Vb, KVb);
    z_kernel<<<(int)((TOK * NH + 255) / 256), 256, 0, stream>>>(Qf, Ksum, Zv, (int)(TOK * NH));

    // 5. attn = (Q @ KV) * Z
    attn_wmma_kernel<<<dim3(LL / 64, BB * NH), 128, 0, stream>>>(Qf, KVb, Zv, Attn);

    // 6. output projection + residual (f32)
    gemm_wmma_kernel<ACT_NONE, 1, 0><<<gE, 256, 0, stream>>>(Attn, EE, WoT, bo, x, x2, EE, EE);

    // 7. FFN block
    ln_bf16_kernel<<<(int)TOK, 256, 0, stream>>>(x2, ln2g, ln2b, h2bf);
    dim3 gF(FF / 128, (int)(TOK / 128));
    gemm_wmma_kernel<ACT_RELU, 0, 1><<<gF, 256, 0, stream>>>(h2bf, EE, W1T, b1, nullptr, h1bf, FF, EE);
    gemm_wmma_kernel<ACT_NONE, 1, 0><<<gE, 256, 0, stream>>>(h1bf, FF, W2T, b2, x2, (float*)d_out, EE, FF);
}
